// RobertaSelfAttention_84971632984561
// MI455X (gfx1250) — compile-verified, matrix operand order checked by reading, not run on hardware
//
#include <hip/hip_runtime.h>

// ---------------------------------------------------------------------------
// RoBERTa self-attention for MI455X (gfx1250, wave32, WMMA bf16 16x16x32).
// - Transpose-free flash attention: S^T = K*Q^T so the score C-fragment is
//   per-lane identical to the B-fragment of P^T needed for ctx^T = V^T*P^T.
// - K/V tiles staged in LDS (double buffered, padded strides) and shared by
//   4 waves x 32 queries = 128 queries per block  -> ~8x less L2 traffic.
// - W slab staged in LDS for the QKV GEMM.
// - Staging uses gfx1250 async global->LDS copies when available.
// ---------------------------------------------------------------------------

typedef __attribute__((ext_vector_type(16))) __bf16          v16bf;
typedef __attribute__((ext_vector_type(8)))  float           v8f;
typedef __attribute__((ext_vector_type(16))) unsigned short  us16;
typedef __attribute__((ext_vector_type(8)))  unsigned short  us8;
typedef __attribute__((ext_vector_type(4)))  int             v4i;

#define S_LEN 4096
#define HID   768
#define NH    12
#define HD    64

#ifndef __has_builtin
#define __has_builtin(x) 0
#endif
#if __has_builtin(__builtin_amdgcn_global_load_async_to_lds_b128) && \
    __has_builtin(__builtin_amdgcn_s_wait_asynccnt)
#define HAVE_ASYNC 1
#else
#define HAVE_ASYNC 0
#endif

// ---- helpers --------------------------------------------------------------

__device__ __forceinline__ unsigned short f32_to_bf16(float f) {
  unsigned int u = __builtin_bit_cast(unsigned int, f);
  u += 0x7fffu + ((u >> 16) & 1u);          // round-to-nearest-even
  return (unsigned short)(u >> 16);
}

// One 16-bit WMMA fragment (A-layout == B-layout pattern on CDNA5):
// lane L owns row/col (L&15); elements 0..7 from `col`, 8..15 from col+16
// (caller folds the lane-half +8 offset into `col`).
__device__ __forceinline__ v16bf load_frag(const unsigned short* base,
                                           int row, int stride, int col) {
  const unsigned short* p = base + (size_t)row * (size_t)stride + col;
  us8 a = *(const us8*)(p);
  us8 b = *(const us8*)(p + 16);
  us16 r = __builtin_shufflevector(a, b, 0,1,2,3,4,5,6,7,8,9,10,11,12,13,14,15);
  return __builtin_bit_cast(v16bf, r);
}

__device__ __forceinline__ v8f wmma_bf16(v16bf a, v16bf b, v8f c) {
  return __builtin_amdgcn_wmma_f32_16x16x32_bf16(false, a, false, b,
                                                 (short)0, c, false, false);
}

// 16-byte global -> LDS copy (async on gfx1250 when the builtin exists).
__device__ __forceinline__ void copy16(const unsigned short* g, unsigned short* l) {
#if HAVE_ASYNC
  __builtin_amdgcn_global_load_async_to_lds_b128(
      (__attribute__((address_space(1))) v4i*)g,
      (__attribute__((address_space(3))) v4i*)l, 0, 0);
#else
  *(us8*)l = *(const us8*)g;
#endif
}

__device__ __forceinline__ void wait_async() {
#if HAVE_ASYNC
  __builtin_amdgcn_s_wait_asynccnt(0);
#endif
}

// ---- conversion kernels ---------------------------------------------------

__global__ void cvt_x_kernel(const float* __restrict__ x,
                             unsigned short* __restrict__ xb, int n) {
  int i = blockIdx.x * blockDim.x + threadIdx.x;
  if (i < n) xb[i] = f32_to_bf16(x[i]);
}

// Wt[m][out][in] = W_m[in][out]  (bf16)
__global__ void cvt_w_kernel(const float* __restrict__ wq,
                             const float* __restrict__ wk,
                             const float* __restrict__ wv,
                             unsigned short* __restrict__ wt) {
  int i = blockIdx.x * blockDim.x + threadIdx.x;
  if (i >= HID * HID) return;
  int m = blockIdx.y;
  const float* w = (m == 0) ? wq : ((m == 1) ? wk : wv);
  int o  = i / HID;
  int in = i - o * HID;
  wt[(size_t)m * HID * HID + (size_t)o * HID + in] =
      f32_to_bf16(w[(size_t)in * HID + o]);
}

// ---- QKV projection GEMM --------------------------------------------------
// Block: 4 waves, each 32 rows (s) x 64 cols (one head). The 64x768 W slab
// is staged once in LDS (row stride padded to 776 -> conflict-free b128).
// grid=(32,NH,3), block=128.
#define WPAD 776

__global__ __launch_bounds__(128)
void qkv_gemm_kernel(const unsigned short* __restrict__ xb,
                     const unsigned short* __restrict__ wt,
                     const float* __restrict__ bq,
                     const float* __restrict__ bk,
                     const float* __restrict__ bv,
                     unsigned short* __restrict__ q,
                     unsigned short* __restrict__ k,
                     unsigned short* __restrict__ vt) {
  __shared__ unsigned short wslab[HD * WPAD];   // ~97 KB of the 320 KB LDS

  const int tid   = threadIdx.x;
  const int lane  = tid & 31;
  const int wid   = tid >> 5;
  const int h     = blockIdx.y;
  const int m     = blockIdx.z;
  const int lr    = lane & 15;
  const int half8 = (lane < 16) ? 0 : 8;
  const int s0    = (blockIdx.x * 4 + wid) * 32;

  // stage W slab: 64 rows x 768 cols = 6144 16B-chunks (64*96 chunks)
  const unsigned short* wm = wt + (size_t)m * HID * HID + (size_t)h * HD * HID;
  for (int c = tid; c < HD * (HID / 8); c += 128) {
    int o  = c / (HID / 8);
    int kc = (c - o * (HID / 8)) * 8;
    copy16(wm + (size_t)o * HID + kc, &wslab[o * WPAD + kc]);
  }
  wait_async();
  __syncthreads();

  const float* bias = (m == 0) ? bq : ((m == 1) ? bk : bv);

  v8f zero = {0.f,0.f,0.f,0.f,0.f,0.f,0.f,0.f};
  v8f acc[2][4] = {{zero, zero, zero, zero}, {zero, zero, zero, zero}};

  for (int kk = 0; kk < HID; kk += 32) {
    v16bf a0 = load_frag(xb, s0 +      lr, HID, kk + half8);
    v16bf a1 = load_frag(xb, s0 + 16 + lr, HID, kk + half8);
#pragma unroll
    for (int dt = 0; dt < 4; ++dt) {
      v16bf b = load_frag(wslab, dt * 16 + lr, WPAD, kk + half8);
      acc[0][dt] = wmma_bf16(a0, b, acc[0][dt]);
      acc[1][dt] = wmma_bf16(a1, b, acc[1][dt]);
    }
  }

#pragma unroll
  for (int st = 0; st < 2; ++st) {
#pragma unroll
    for (int dt = 0; dt < 4; ++dt) {
      const int col = h * HD + dt * 16 + lr;
      v8f r = acc[st][dt] + bias[col];
      if (m == 0) r = r * 0.125f;               // fold 1/sqrt(HD) into Q
      if (m < 2) {                              // q/k: [h][s][d] bf16
        unsigned short* dst = (m == 0) ? q : k;
#pragma unroll
        for (int j = 0; j < 8; ++j) {
          int s = s0 + st * 16 + half8 + j;
          dst[((size_t)h * S_LEN + s) * HD + dt * 16 + lr] = f32_to_bf16(r[j]);
        }
      } else {                                  // v: transposed [h][d][s] bf16
        us8 hv;
#pragma unroll
        for (int j = 0; j < 8; ++j) hv[j] = f32_to_bf16(r[j]);
        *(us8*)(vt + ((size_t)h * HD + dt * 16 + lr) * S_LEN
                   + s0 + st * 16 + half8) = hv;
      }
    }
  }
}

// ---- flash attention ------------------------------------------------------
// Block: 4 waves x 32 queries = 128 queries of one head; K/V 32-key tiles
// staged in double-buffered LDS (padded strides: K rows 72, V rows 40).
// grid=(32,NH), block=128.
#define KPAD 72
#define VPAD 40

__global__ __launch_bounds__(128)
void attn_kernel(const unsigned short* __restrict__ q,
                 const unsigned short* __restrict__ k,
                 const unsigned short* __restrict__ vt,
                 const float* __restrict__ mask,
                 float* __restrict__ out) {
  __shared__ unsigned short ktile[2][32 * KPAD];
  __shared__ unsigned short vtile[2][HD * VPAD];

  const int tid   = threadIdx.x;
  const int lane  = tid & 31;
  const int wid   = tid >> 5;
  const int h     = blockIdx.y;
  const int q0    = (blockIdx.x * 4 + wid) * 32;
  const int lr    = lane & 15;
  const int half8 = (lane < 16) ? 0 : 8;

  const unsigned short* qh = q  + (size_t)h * S_LEN * HD;
  const unsigned short* kh = k  + (size_t)h * S_LEN * HD;
  const unsigned short* vh = vt + (size_t)h * HD * S_LEN;

  // Q^T B-fragments: 2 query sub-tiles x 2 head-dim chunks
  v16bf bqf[2][2];
#pragma unroll
  for (int qs = 0; qs < 2; ++qs) {
    bqf[qs][0] = load_frag(qh, q0 + qs * 16 + lr, HD,  0 + half8);
    bqf[qs][1] = load_frag(qh, q0 + qs * 16 + lr, HD, 32 + half8);
  }

  v8f zero = {0.f,0.f,0.f,0.f,0.f,0.f,0.f,0.f};
  v8f acc[2][4] = {{zero, zero, zero, zero}, {zero, zero, zero, zero}};
  float mrun[2] = {-__builtin_inff(), -__builtin_inff()};
  float lrun[2] = {0.f, 0.f};

  // ---- stage helper: 32x64 K tile + 64x32 V^T tile -> LDS ----
  auto stage = [&](int buf, int k0) {
#pragma unroll
    for (int rr = 0; rr < 2; ++rr) {
      int c = tid + rr * 128;                      // 0..255
      int krow = c >> 3, kcol = (c & 7) * 8;
      copy16(kh + (size_t)(k0 + krow) * HD + kcol,
             &ktile[buf][krow * KPAD + kcol]);
      int vrow = c >> 2, vcol = (c & 3) * 8;
      copy16(vh + (size_t)vrow * S_LEN + k0 + vcol,
             &vtile[buf][vrow * VPAD + vcol]);
    }
  };

  stage(0, 0);
  int cur = 0;

  for (int k0 = 0; k0 < S_LEN; k0 += 32) {
    wait_async();
    __syncthreads();                 // ktile[cur]/vtile[cur] ready for all
    const int nxt = cur ^ 1;
    if (k0 + 32 < S_LEN) stage(nxt, k0 + 32);   // overlap with compute

    const unsigned short* kt = ktile[cur];
    const unsigned short* vs = vtile[cur];

    // K A-fragments (shared by both query sub-tiles)
    v16bf ka00 = load_frag(kt,      lr, KPAD,  0 + half8);
    v16bf ka01 = load_frag(kt,      lr, KPAD, 32 + half8);
    v16bf ka10 = load_frag(kt, 16 + lr, KPAD,  0 + half8);
    v16bf ka11 = load_frag(kt, 16 + lr, KPAD, 32 + half8);
    // V^T A-fragments
    v16bf va[4];
#pragma unroll
    for (int dt = 0; dt < 4; ++dt)
      va[dt] = load_frag(vs, dt * 16 + lr, VPAD, half8);

    v8f m0 = *(const v8f*)(mask + k0 +      half8);
    v8f m1 = *(const v8f*)(mask + k0 + 16 + half8);

#pragma unroll
    for (int qs = 0; qs < 2; ++qs) {
      // S^T = K * Q^T : two 16-key row tiles
      v8f t0 = wmma_bf16(ka00, bqf[qs][0], zero);
      t0     = wmma_bf16(ka01, bqf[qs][1], t0);
      v8f t1 = wmma_bf16(ka10, bqf[qs][0], zero);
      t1     = wmma_bf16(ka11, bqf[qs][1], t1);
      t0 = t0 + m0;
      t1 = t1 + m1;

      // online softmax: in-lane over 16 keys + one xor-16 shuffle
      float mloc = t0[0];
#pragma unroll
      for (int j = 1; j < 8; ++j) mloc = fmaxf(mloc, t0[j]);
#pragma unroll
      for (int j = 0; j < 8; ++j) mloc = fmaxf(mloc, t1[j]);
      float mtile = fmaxf(mloc, __shfl_xor(mloc, 16, 32));
      float mnew  = fmaxf(mrun[qs], mtile);
      float alpha = __expf(mrun[qs] - mnew);

      float p0[8], p1[8];
      float lloc = 0.f;
#pragma unroll
      for (int j = 0; j < 8; ++j) { p0[j] = __expf(t0[j] - mnew); lloc += p0[j]; }
#pragma unroll
      for (int j = 0; j < 8; ++j) { p1[j] = __expf(t1[j] - mnew); lloc += p1[j]; }
      float ltile = lloc + __shfl_xor(lloc, 16, 32);
      lrun[qs] = lrun[qs] * alpha + ltile;
      mrun[qs] = mnew;

      // P^T B-fragment straight from the S^T C-fragment (no lane movement)
      us16 pu;
#pragma unroll
      for (int j = 0; j < 8; ++j) {
        pu[j]     = f32_to_bf16(p0[j]);
        pu[8 + j] = f32_to_bf16(p1[j]);
      }
      v16bf pb = __builtin_bit_cast(v16bf, pu);

#pragma unroll
      for (int dt = 0; dt < 4; ++dt) {
        acc[qs][dt] = acc[qs][dt] * alpha;
        acc[qs][dt] = wmma_bf16(va[dt], pb, acc[qs][dt]);
      }
    }
    cur = nxt;
  }

#pragma unroll
  for (int qs = 0; qs < 2; ++qs) {
    const float inv = 1.0f / lrun[qs];
    const int s = q0 + qs * 16 + lr;
    float* op = out + (size_t)s * HID + h * HD + half8;
    *(v8f*)(op +  0) = acc[qs][0] * inv;
    *(v8f*)(op + 16) = acc[qs][1] * inv;
    *(v8f*)(op + 32) = acc[qs][2] * inv;
    *(v8f*)(op + 48) = acc[qs][3] * inv;
  }
}

// ---- launcher -------------------------------------------------------------

extern "C" void kernel_launch(void* const* d_in, const int* in_sizes, int n_in,
                              void* d_out, int out_size, void* d_ws, size_t ws_size,
                              hipStream_t stream) {
  const float* hidden = (const float*)d_in[0];
  const float* amask  = (const float*)d_in[1];
  const float* Wq     = (const float*)d_in[2];
  const float* bq     = (const float*)d_in[3];
  const float* Wk     = (const float*)d_in[4];
  const float* bk     = (const float*)d_in[5];
  const float* Wv     = (const float*)d_in[6];
  const float* bv     = (const float*)d_in[7];
  float* out = (float*)d_out;

  // workspace layout (bytes)
  char* ws = (char*)d_ws;
  unsigned short* xb = (unsigned short*)(ws);                          // 4096*768
  unsigned short* wt = (unsigned short*)(ws + 6291456);                // 3*768*768
  unsigned short* qb = (unsigned short*)(ws + 6291456 + 3538944);      // 12*4096*64
  unsigned short* kb = (unsigned short*)(ws + 6291456 + 3538944 + 6291456);
  unsigned short* vtb= (unsigned short*)(ws + 6291456 + 3538944 + 2*6291456);

  const int nX = S_LEN * HID;
  cvt_x_kernel<<<(nX + 255) / 256, 256, 0, stream>>>(hidden, xb, nX);
  cvt_w_kernel<<<dim3((HID * HID + 255) / 256, 3), 256, 0, stream>>>(Wq, Wk, Wv, wt);

  qkv_gemm_kernel<<<dim3(S_LEN / 128, NH, 3), 128, 0, stream>>>(
      xb, wt, bq, bk, bv, qb, kb, vtb);

  attn_kernel<<<dim3(S_LEN / 128, NH), 128, 0, stream>>>(qb, kb, vtb, amask, out);
}
